// MultiHeadSelfAttentionEinSum_8022998909676
// MI455X (gfx1250) — matrix operand order repaired and checked by reading, not compiled, not run on hardware
//
#include <hip/hip_runtime.h>

// ---------------------------------------------------------------------------
// MHA for MI455X (gfx1250): wave32, v_wmma_f32_16x16x32_f16, async-to-LDS.
// B=4, T=2048, E=1024, N=16 heads, P=64, SCALE = 8 -> 1/SCALE = 0.125 exact.
// One-time f16 conversion/transpose of x and weights (all L2-resident) so the
// GEMM hot loops issue only contiguous b128 f16 loads + WMMAs.
// ---------------------------------------------------------------------------

typedef __attribute__((ext_vector_type(16))) _Float16 v16h;
typedef __attribute__((ext_vector_type(8)))  _Float16 v8h;
typedef __attribute__((ext_vector_type(8)))  float    v8f;

#define WMMA_F16(A, B, C) \
  __builtin_amdgcn_wmma_f32_16x16x32_f16(false, (A), false, (B), (short)0, (C), false, false)

static __device__ __forceinline__ float halfgrp_max16(float v) {
  v = fmaxf(v, __shfl_xor(v, 1, 32));
  v = fmaxf(v, __shfl_xor(v, 2, 32));
  v = fmaxf(v, __shfl_xor(v, 4, 32));
  v = fmaxf(v, __shfl_xor(v, 8, 32));
  return v;
}
static __device__ __forceinline__ float halfgrp_sum16(float v) {
  v += __shfl_xor(v, 1, 32);
  v += __shfl_xor(v, 2, 32);
  v += __shfl_xor(v, 4, 32);
  v += __shfl_xor(v, 8, 32);
  return v;
}

// ---------------------------------------------------------------------------
// Prep kernels: one-time fp32 -> f16 conversion (+ transpose for weights).
// ---------------------------------------------------------------------------
__global__ void convert_x_kernel(const float* __restrict__ x,
                                 _Float16* __restrict__ xh) {
  const size_t i = ((size_t)blockIdx.x * 256 + threadIdx.x) * 4;
  const float4 v = *(const float4*)(x + i);
  xh[i + 0] = (_Float16)v.x;
  xh[i + 1] = (_Float16)v.y;
  xh[i + 2] = (_Float16)v.z;
  xh[i + 3] = (_Float16)v.w;
}

// w[K][C] fp32 -> wT[C][K] f16 (writes coalesced over k)
__global__ void transpose_w_kernel(const float* __restrict__ w,
                                   _Float16* __restrict__ wT,
                                   int K, int C) {
  const size_t idx = (size_t)blockIdx.x * 256 + threadIdx.x;  // over K*C
  const int c = (int)(idx / K);
  const int k = (int)(idx % K);
  wT[idx] = (_Float16)w[(size_t)k * C + c];
}

// ---------------------------------------------------------------------------
// Kernel 1: QKV projection.  Xh[8192,1024]f16 x WqkvT[3072,1024]f16 + bias.
// Each wave: 64x32 strip (4 M x 2 N subtiles, 8 accumulators).
// ---------------------------------------------------------------------------
__global__ void qkv_proj_kernel(const _Float16* __restrict__ xh,
                                const _Float16* __restrict__ wT,
                                const float* __restrict__ bias,
                                _Float16* __restrict__ qb,
                                _Float16* __restrict__ kbuf,
                                _Float16* __restrict__ vbuf) {
  const int lane = threadIdx.x & 31;
  const int wav  = threadIdx.x >> 5;
  const int strip = blockIdx.x * 8 + wav;         // 128 * 96 strips
  const int n0 = (strip % 96) * 32;
  const int m0 = (strip / 96) * 64;
  const int r  = lane & 15;
  const int hf = lane >> 4;
  const _Float16* arow  = xh + (size_t)(m0 + r) * 1024;
  const _Float16* bcol0 = wT + (size_t)(n0 + r) * 1024;
  const _Float16* bcol1 = wT + (size_t)(n0 + 16 + r) * 1024;

  v8f acc[4][2] = {};
  for (int k0 = 0; k0 < 1024; k0 += 32) {
    const int kb = k0 + hf * 16;                  // B-layout K base
    v16h bf0, bf1;
    {
      const v8h b0lo = *(const v8h*)(bcol0 + kb);
      const v8h b0hi = *(const v8h*)(bcol0 + kb + 8);
      const v8h b1lo = *(const v8h*)(bcol1 + kb);
      const v8h b1hi = *(const v8h*)(bcol1 + kb + 8);
#pragma unroll
      for (int i = 0; i < 8; ++i) {
        bf0[i] = b0lo[i]; bf0[8 + i] = b0hi[i];
        bf1[i] = b1lo[i]; bf1[8 + i] = b1hi[i];
      }
    }
    const int ka = k0 + hf * 16;                  // A-layout K base
#pragma unroll
    for (int s = 0; s < 4; ++s) {
      const _Float16* ar = arow + (size_t)(s * 16) * 1024;
      const v8h alo = *(const v8h*)(ar + ka);
      const v8h ahi = *(const v8h*)(ar + ka + 8);
      v16h af;
#pragma unroll
      for (int i = 0; i < 8; ++i) { af[i] = alo[i]; af[8 + i] = ahi[i]; }
      acc[s][0] = WMMA_F16(af, bf0, acc[s][0]);
      acc[s][1] = WMMA_F16(af, bf1, acc[s][1]);
    }
  }

  // epilogue: bias + scatter f16 to Q/K/V [B,N,T,P].  col fixed per lane.
#pragma unroll
  for (int nt = 0; nt < 2; ++nt) {
    const int col = n0 + nt * 16 + r;             // [0,3072)
    const float bv = bias[col];
    const int n  = col / 192;
    const int j  = col % 192;
    const int wh = j >> 6;                        // 0=Q 1=K 2=V
    const int p  = j & 63;
    _Float16* dst = (wh == 0) ? qb : (wh == 1) ? kbuf : vbuf;
#pragma unroll
    for (int s = 0; s < 4; ++s)
#pragma unroll
      for (int i = 0; i < 8; ++i) {
        const int m = m0 + s * 16 + hf * 8 + i;
        dst[(((size_t)((m >> 11) * 16 + n) * 2048) + (m & 2047)) * 64 + p] =
            (_Float16)(acc[s][nt][i] + bv);
      }
  }
}

// ---------------------------------------------------------------------------
// Kernel 2: flash attention.  One block = 8 waves = 128 query rows of one
// (batch, head); 32-key K/V strips staged into LDS once per block via
// global_load_async_to_lds_b128 (ASYNCcnt) and shared by all 8 waves.
// ---------------------------------------------------------------------------
__global__ void flash_attn_kernel(const _Float16* __restrict__ qb,
                                  const _Float16* __restrict__ kbuf,
                                  const _Float16* __restrict__ vbuf,
                                  _Float16* __restrict__ wtd) {
  __shared__ _Float16 sK[32 * 64];                // 4 KB: 32 keys x P
  __shared__ _Float16 sV[32 * 64];                // 4 KB
  __shared__ _Float16 sP[8][512];                 // per-wave 16x32 exp(S)

  const int tid  = threadIdx.x;
  const int lane = tid & 31;
  const int wav  = tid >> 5;
  const int head    = (blockIdx.x >> 4) & 15;
  const int batch   = blockIdx.x >> 8;
  const int rowTile = (blockIdx.x & 15) * 8 + wav;
  const int r  = lane & 15;
  const int hf = lane >> 4;
  const size_t bh = (size_t)(batch * 16 + head) * 2048;
  const int t0 = rowTile * 16;

  // cooperative staging: 256 threads x 16B covers one 32x64 f16 strip
  const int srow = tid >> 3;                      // 0..31 (key row)
  const int schk = (tid & 7) * 8;                 // half-element chunk
  const unsigned ldsK = (unsigned)(size_t)&sK[srow * 64 + schk];
  const unsigned ldsV = (unsigned)(size_t)&sV[srow * 64 + schk];
  const _Float16* gk = kbuf + bh * 64;
  const _Float16* gv = vbuf + bh * 64;

  // Q A-fragments (Kdim = P = 64 -> two 16x32 fragments), loaded once.
  v16h qa0, qa1;
  const _Float16* qrow = qb + (bh + t0 + r) * 64;
#pragma unroll
  for (int i = 0; i < 8; ++i) {
    qa0[i]     = qrow[hf * 16 + i];
    qa0[8 + i] = qrow[hf * 16 + 8 + i];
    qa1[i]     = qrow[32 + hf * 16 + i];
    qa1[8 + i] = qrow[32 + hf * 16 + 8 + i];
  }

  float Mrun[8], Lrun[8];
#pragma unroll
  for (int i = 0; i < 8; ++i) { Mrun[i] = -3.0e38f; Lrun[i] = 0.f; }
  v8f O[4] = {};

  for (int j0 = 0; j0 < 2048; j0 += 32) {
    __syncthreads();                              // previous strip readers done
    const unsigned goff = (unsigned)((((j0 + srow) * 64) + schk) * 2);
    asm volatile("global_load_async_to_lds_b128 %0, %1, %2 offset:0"
                 :: "v"(ldsK), "v"(goff), "s"(gk) : "memory");
    asm volatile("global_load_async_to_lds_b128 %0, %1, %2 offset:0"
                 :: "v"(ldsV), "v"(goff), "s"(gv) : "memory");
    asm volatile("s_wait_asynccnt 0" ::: "memory");
    __syncthreads();                              // strip visible to all waves

    // ---- scores: two 16x16 D tiles over the 32-key strip -----------------
    v8f s0 = {}, s1 = {};
    {
      const _Float16* k0p = &sK[r * 64];          // column = key r
      const _Float16* k1p = &sK[(16 + r) * 64];   // column = key 16+r
      v16h b0, b1;
#pragma unroll
      for (int i = 0; i < 16; ++i) { b0[i] = k0p[hf * 16 + i]; b1[i] = k1p[hf * 16 + i]; }
      s0 = WMMA_F16(qa0, b0, s0);
      s1 = WMMA_F16(qa0, b1, s1);
#pragma unroll
      for (int i = 0; i < 16; ++i) { b0[i] = k0p[32 + hf * 16 + i]; b1[i] = k1p[32 + hf * 16 + i]; }
      s0 = WMMA_F16(qa1, b0, s0);
      s1 = WMMA_F16(qa1, b1, s1);
    }

    // ---- online softmax (rows = i + 8*hf, cols across 16-lane halves) ----
    float p0[8], p1[8];
#pragma unroll
    for (int i = 0; i < 8; ++i) {
      const float v0 = s0[i] * 0.125f;            // 1/sqrt(P)
      const float v1 = s1[i] * 0.125f;
      const float rm   = halfgrp_max16(fmaxf(v0, v1));
      const float newM = fmaxf(Mrun[i], rm);
      const float corr = __expf(Mrun[i] - newM);
      Mrun[i] = newM;
      p0[i] = __expf(v0 - newM);
      p1[i] = __expf(v1 - newM);
      const float rs = halfgrp_sum16(p0[i] + p1[i]);
      Lrun[i] = Lrun[i] * corr + rs;
      O[0][i] *= corr; O[1][i] *= corr; O[2][i] *= corr; O[3][i] *= corr;
    }

    // ---- D-layout -> A-layout reshape through per-wave LDS ----------------
#pragma unroll
    for (int i = 0; i < 8; ++i) {
      const int rr = hf * 8 + i;
      sP[wav][rr * 32 + r]      = (_Float16)p0[i];
      sP[wav][rr * 32 + 16 + r] = (_Float16)p1[i];
    }
    v16h pa;
#pragma unroll
    for (int i = 0; i < 8; ++i) {
      pa[i]     = sP[wav][r * 32 + hf * 8 + i];
      pa[8 + i] = sP[wav][r * 32 + hf * 8 + 16 + i];
    }

    // ---- O += P x V : 4 slices of the P=64 dimension ----------------------
#pragma unroll
    for (int ps = 0; ps < 4; ++ps) {
      const int pcol = ps * 16 + r;
      v16h vB;
#pragma unroll
      for (int i = 0; i < 16; ++i)
        vB[i] = sV[(hf * 8 + i + (i & 8)) * 64 + pcol];
      O[ps] = WMMA_F16(pa, vB, O[ps]);
    }
  }

  // ---- normalize and store weighted as [B*T, c = p*N + head] f16 ----------
#pragma unroll
  for (int i = 0; i < 8; ++i) {
    const float invL = 1.0f / Lrun[i];
    const size_t mrow = (size_t)(batch * 2048 + t0 + hf * 8 + i) * 1024;
#pragma unroll
    for (int ps = 0; ps < 4; ++ps)
      wtd[mrow + (ps * 16 + r) * 16 + head] = (_Float16)(O[ps][i] * invL);
  }
}

// ---------------------------------------------------------------------------
// Kernel 3: output projection.  Wtd[8192,1024]f16 x WoT[1024,1024]f16 + b_o
// -> fp32.  Same 64x32 strip-per-wave structure as kernel 1.
// ---------------------------------------------------------------------------
__global__ void out_proj_kernel(const _Float16* __restrict__ wtd,
                                const _Float16* __restrict__ woT,
                                const float* __restrict__ bo,
                                float* __restrict__ out) {
  const int lane = threadIdx.x & 31;
  const int wav  = threadIdx.x >> 5;
  const int strip = blockIdx.x * 8 + wav;         // 128 * 32 strips
  const int n0 = (strip % 32) * 32;
  const int m0 = (strip / 32) * 64;
  const int r  = lane & 15;
  const int hf = lane >> 4;
  const _Float16* arow  = wtd + (size_t)(m0 + r) * 1024;
  const _Float16* bcol0 = woT + (size_t)(n0 + r) * 1024;
  const _Float16* bcol1 = woT + (size_t)(n0 + 16 + r) * 1024;

  v8f acc[4][2] = {};
  for (int k0 = 0; k0 < 1024; k0 += 32) {
    const int kb = k0 + hf * 16;
    v16h bf0, bf1;
    {
      const v8h b0lo = *(const v8h*)(bcol0 + kb);
      const v8h b0hi = *(const v8h*)(bcol0 + kb + 8);
      const v8h b1lo = *(const v8h*)(bcol1 + kb);
      const v8h b1hi = *(const v8h*)(bcol1 + kb + 8);
#pragma unroll
      for (int i = 0; i < 8; ++i) {
        bf0[i] = b0lo[i]; bf0[8 + i] = b0hi[i];
        bf1[i] = b1lo[i]; bf1[8 + i] = b1hi[i];
      }
    }
    const int ka = k0 + hf * 16;
#pragma unroll
    for (int s = 0; s < 4; ++s) {
      const _Float16* ar = arow + (size_t)(s * 16) * 1024;
      const v8h alo = *(const v8h*)(ar + ka);
      const v8h ahi = *(const v8h*)(ar + ka + 8);
      v16h af;
#pragma unroll
      for (int i = 0; i < 8; ++i) { af[i] = alo[i]; af[8 + i] = ahi[i]; }
      acc[s][0] = WMMA_F16(af, bf0, acc[s][0]);
      acc[s][1] = WMMA_F16(af, bf1, acc[s][1]);
    }
  }

#pragma unroll
  for (int nt = 0; nt < 2; ++nt) {
    const int col = n0 + nt * 16 + r;
    const float bv = bo[col];
#pragma unroll
    for (int s = 0; s < 4; ++s)
#pragma unroll
      for (int i = 0; i < 8; ++i)
        out[(size_t)(m0 + s * 16 + hf * 8 + i) * 1024 + col] = acc[s][nt][i] + bv;
  }
}

// ---------------------------------------------------------------------------
extern "C" void kernel_launch(void* const* d_in, const int* in_sizes, int n_in,
                              void* d_out, int out_size, void* d_ws, size_t ws_size,
                              hipStream_t stream) {
  (void)in_sizes; (void)n_in; (void)out_size; (void)ws_size;
  const float* x     = (const float*)d_in[0];   // [B,T,E]
  const float* w_qkv = (const float*)d_in[1];   // [E,3072]
  const float* b_qkv = (const float*)d_in[2];   // [3072]
  const float* w_o   = (const float*)d_in[3];   // [1024,1024] (c = p*N+n major)
  const float* b_o   = (const float*)d_in[4];   // [E]
  float* out = (float*)d_out;                   // [B,T,E]

  const size_t NQ = (size_t)4 * 16 * 2048 * 64; // 8388608 elems per tensor
  _Float16* q     = (_Float16*)d_ws;
  _Float16* k     = q + NQ;
  _Float16* v     = k + NQ;
  _Float16* xh    = v + NQ;                     // x in f16 (dead after GEMM1)
  _Float16* wt    = xh;                         // attention output reuses xh
  _Float16* wqkvT = xh + NQ;                    // [3072][1024] f16
  _Float16* woT   = wqkvT + (size_t)3072 * 1024;// [1024][1024] f16

  convert_x_kernel  <<< 8192, 256, 0, stream>>>(x, xh);
  transpose_w_kernel<<<12288, 256, 0, stream>>>(w_qkv, wqkvT, 1024, 3072);
  transpose_w_kernel<<< 4096, 256, 0, stream>>>(w_o,   woT,   1024, 1024);
  qkv_proj_kernel   <<< 1536, 256, 0, stream>>>(xh, wqkvT, b_qkv, q, k, v);
  flash_attn_kernel <<< 1024, 256, 0, stream>>>(q, k, v, wt);
  out_proj_kernel   <<<  512, 256, 0, stream>>>(wt, woT, b_o, out);
}
